// RefPhi4FlashDiffAttention_56650618635017
// MI455X (gfx1250) — matrix operand order repaired and checked by reading, not compiled, not run on hardware
//
#include <hip/hip_runtime.h>

#define DEV static __device__ __forceinline__

typedef __attribute__((ext_vector_type(16))) __bf16 v16bf;
typedef __attribute__((ext_vector_type(8))) float v8f;
typedef __attribute__((ext_vector_type(8))) unsigned v8u;

union LdFrag {      // load-side only: LDS b128 pairs -> WMMA operand
  v16bf bf;
  uint4 q[2];
};

constexpr int kB = 2, kS = 2048, kH = 32, kHP = 16, kD = 128;
constexpr int QT = 64;   // query rows per block
constexpr int KT = 32;   // keys per step
// softmax in exp2 domain: scale = 1/sqrt(128) * log2(e)
constexpr float kScaleL2 = 0.08838834764831845f * 1.4426950408889634f;
constexpr float kLambdaInit = 0.7836057662568868f; // 0.8 - 0.6*exp(-3.6)
constexpr float kNegInf = -1.0e30f;

// ---- LDS layout (bytes). Fragment loads are all 16B aligned. ----
constexpr int KSTR = 136;           // ushorts per K row (128 + 8 pad); 272B row, 16B-mult
constexpr int K1_OFF = 0;           // 32*272 = 8704 B
constexpr int K2_OFF = 8704;        // 8704 B
constexpr int VSTR = 40;            // ushorts per VT row (32 + 8 pad); 80B, 16B-mult
constexpr int VT_OFF = 17408;       // 256*80 = 20480 B -> ends 37888
constexpr int XSTR = 264;           // floats per xbuf row (256 + 8 pad)
constexpr int SMEM_BYTES = 64 * XSTR * 4; // 67584; xbuf reuses K/V space after loop

#if __has_builtin(__builtin_amdgcn_cvt_pk_bf16_f32)
typedef __attribute__((ext_vector_type(2))) __bf16 v2bf;
DEV unsigned pack2(float a, float b) {
  v2bf r = __builtin_amdgcn_cvt_pk_bf16_f32(a, b);
  return __builtin_bit_cast(unsigned, r);
}
#else
// 3-op bf16 pair pack: round-half-up (matches RNE except exact ties), then
// one v_perm_b32 grabs both high halves. Stays in 32-bit int domain so the
// backend never materializes 16-bit element shuffles.
DEV unsigned pack2(float a, float b) {
  unsigned ua = __float_as_uint(a) + 0x8000u;
  unsigned ub = __float_as_uint(b) + 0x8000u;
  return __builtin_amdgcn_perm(ub, ua, 0x07060302u);
}
#endif

DEV float fexp2(float x) { return __builtin_amdgcn_exp2f(x); }
// raw lane gather (byte index), no bounds-check glue
DEV float bperm_f(int byteidx, float x) {
  return __int_as_float(__builtin_amdgcn_ds_bpermute(byteidx, __float_as_int(x)));
}
// reduction across the 16 lanes holding one output row (epilogue RMSNorm only)
DEV float rowred_sum(float x) {
#pragma unroll
  for (int m = 1; m < 16; m <<= 1) x += __shfl_xor(x, m, 32);
  return x;
}

__global__ __launch_bounds__(256, 1)
void diff_attn_wmma(const float* __restrict__ q, const float* __restrict__ k,
                    const float* __restrict__ v,
                    const float* __restrict__ lq1, const float* __restrict__ lk1,
                    const float* __restrict__ lq2, const float* __restrict__ lk2,
                    const float* __restrict__ w, float* __restrict__ out) {
  extern __shared__ unsigned char smem[];
  unsigned short* kls0 = (unsigned short*)(smem + K1_OFF);
  unsigned short* kls1 = (unsigned short*)(smem + K2_OFF);
  unsigned short* vtl = (unsigned short*)(smem + VT_OFF);
  float* xbuf = (float*)(smem);

  const int tid = threadIdx.x;
  const int lane = tid & 31;
  const int wv = tid >> 5;    // 0..7
  const int att = wv >> 2;    // 0: attn1, 1: attn2
  const int qr = wv & 3;      // 16-row group within the 64-row tile
  const int qt = blockIdx.x, hp = blockIdx.y, b = blockIdx.z;
  const int qbase = qt * QT;
  const int r0 = qbase + 16 * qr;
  const int half = lane >> 4;
  const int lm = lane & 15;

  // lane-gather byte indices: cross-half partner, and per-O-row stat broadcast
  const int xidx = (lane ^ 16) * 4;
  int bidx[8];
#pragma unroll
  for (int vv = 0; vv < 8; ++vv) bidx[vv] = 4 * (vv + ((lane & 16) >> 1));

  // lambda scalar (tiny, computed redundantly per thread; uniform -> scalar loads)
  float d1 = 0.f, d2 = 0.f;
  for (int i = 0; i < kD; ++i) { d1 += lq1[i] * lk1[i]; d2 += lq2[i] * lk2[i]; }
  const float lam = __expf(d1) - __expf(d2) + kLambdaInit;

  // ---- Q as B-fragments: lane = N(query) = lm, elements = K(d) = t*32 + 8*half + (e<8 ? e : e+8) ----
  v16bf qb[4];
  {
    const float* qrow = q + ((b * kS + r0 + lm) * kH + 2 * hp + att) * kD;
#pragma unroll
    for (int t = 0; t < 4; ++t) {
      const int d0 = t * 32 + half * 8;
      v8u tmp;
#pragma unroll
      for (int j = 0; j < 4; ++j) {
        float4 f = *(const float4*)(qrow + d0 + 4 * (j & 1) + 16 * (j >> 1));
        tmp[2 * j + 0] = pack2(f.x, f.y);
        tmp[2 * j + 1] = pack2(f.z, f.w);
      }
      qb[t] = __builtin_bit_cast(v16bf, tmp);
    }
  }

  // fp32 accumulators: 16 N-tiles of [16x16] -> O[16 rows][256 cols]
  v8f o[16];
#pragma unroll
  for (int j = 0; j < 16; ++j)
#pragma unroll
    for (int e = 0; e < 8; ++e) o[j][e] = 0.f;

  float mold = -3.0e38f, lrow = 0.f;  // per-lane: softmax state of query (lane&15)

  // cooperative-load assignments
  const int ldr = tid >> 3;          // K: key row 0..31
  const int ldc = tid & 7;           // K: 16-col chunk
  const int vkp = (tid >> 4) * 2;    // V: key pair 0,2,..,30
  const int vc0 = (tid & 15) * 16;   // V: 16-col chunk

  const int nsteps = (qbase >> 5) + 2; // keys [0, qbase+64) covers causal span

  for (int step = 0; step < nsteps; ++step) {
    const int k0 = step * KT;
    __syncthreads();
    {
      // K1 / K2: 16 cols per thread, bf16 row-major
      const int krbase = ((b * kS + k0 + ldr) * kH + 2 * hp) * kD;
      const int c0 = ldc * 16;
#pragma unroll
      for (int a = 0; a < 2; ++a) {
        const float* src = k + krbase + a * kD + c0;
        unsigned short* dst = (a ? kls1 : kls0) + ldr * KSTR + c0;
#pragma unroll
        for (int j = 0; j < 4; ++j) {
          float4 f = *(const float4*)(src + 4 * j);
          uint2 p; p.x = pack2(f.x, f.y); p.y = pack2(f.z, f.w);
          *(uint2*)(dst + 4 * j) = p;
        }
      }
      // V12: a key-PAIR per thread so each transposed store is one b32
      const int vrbase = ((b * kS + k0 + vkp) * kH + 2 * hp) * kD + vc0;
      const float* vs0 = v + vrbase;
      const float* vs1 = v + vrbase + kH * kD;  // next key row
#pragma unroll
      for (int j = 0; j < 4; ++j) {
        float4 f0 = *(const float4*)(vs0 + 4 * j);
        float4 f1 = *(const float4*)(vs1 + 4 * j);
        const int c = vc0 + 4 * j;
        *(unsigned*)(vtl + (c + 0) * VSTR + vkp) = pack2(f0.x, f1.x);
        *(unsigned*)(vtl + (c + 1) * VSTR + vkp) = pack2(f0.y, f1.y);
        *(unsigned*)(vtl + (c + 2) * VSTR + vkp) = pack2(f0.z, f1.z);
        *(unsigned*)(vtl + (c + 3) * VSTR + vkp) = pack2(f0.w, f1.w);
      }
      if (step + 1 < nsteps) {
        __builtin_prefetch(k + krbase + KT * kH * kD, 0, 1);
        __builtin_prefetch(v + vrbase + KT * kH * kD, 0, 1);
      }
    }
    __syncthreads();

    if (k0 <= r0 + 15) {   // wave-uniform: skip fully-masked tiles (EXEC stays full)
      const unsigned short* kl = att ? kls1 : kls0;
      // ---- transposed scores: S^T[32 keys x 16 queries] = K_tile * Q^T ----
      // software-pipelined: preload A-fragments for t+1 before WMMA t
      v8f sc0, sc1;
#pragma unroll
      for (int e = 0; e < 8; ++e) { sc0[e] = 0.f; sc1[e] = 0.f; }
      const unsigned short* kfp0 = kl + lm * KSTR + half * 8;
      const unsigned short* kfp1 = kl + (16 + lm) * KSTR + half * 8;
      LdFrag ka0, ka1;
      ka0.q[0] = *(const uint4*)(kfp0);
      ka0.q[1] = *(const uint4*)(kfp0 + 16);
      ka1.q[0] = *(const uint4*)(kfp1);
      ka1.q[1] = *(const uint4*)(kfp1 + 16);
#pragma unroll
      for (int t = 0; t < 4; ++t) {
        LdFrag na0, na1;
        if (t < 3) {
          na0.q[0] = *(const uint4*)(kfp0 + 32 * (t + 1));
          na0.q[1] = *(const uint4*)(kfp0 + 32 * (t + 1) + 16);
          na1.q[0] = *(const uint4*)(kfp1 + 32 * (t + 1));
          na1.q[1] = *(const uint4*)(kfp1 + 32 * (t + 1) + 16);
        }
        sc0 = __builtin_amdgcn_wmma_f32_16x16x32_bf16(
            false, ka0.bf, false, qb[t], (short)0, sc0, false, false);
        sc1 = __builtin_amdgcn_wmma_f32_16x16x32_bf16(
            false, ka1.bf, false, qb[t], (short)0, sc1, false, false);
        ka0 = na0; ka1 = na1;
      }
      // ---- online softmax; lane owns query (lane&15), keys k0+8*half+{e, 16+e} ----
      // raw scores kept; scale folded into the reduced max and the exp FMA
      if (k0 + KT - 1 > r0) {         // only the single diagonal step needs masking
        const int qg = r0 + lm;
        const int kb = k0 + 8 * half;
#pragma unroll
        for (int e = 0; e < 8; ++e) {
          if (kb + e > qg) sc0[e] = kNegInf;
          if (kb + 16 + e > qg) sc1[e] = kNegInf;
        }
      }
      float mxr = -3.0e38f;
#pragma unroll
      for (int e = 0; e < 8; ++e) mxr = fmaxf(mxr, fmaxf(sc0[e], sc1[e]));
      float mx = mxr * kScaleL2;                  // into scaled/log2 domain
      mx = fmaxf(mx, bperm_f(xidx, mx));          // combine the two key halves
      const float mnew = fmaxf(mold, mx);
      const bool renew = __any(mx > mold);        // wave-uniform
      // P = exp2(s*scale - m), packed straight into the PV A-fragment:
      // A element e <-> key 8*half + (e<8 ? e : e+8) == sc0[e] / sc1[e-8]
      const float nm = -mnew;
      v8u pau;
      float ls = 0.f;
#pragma unroll
      for (int i = 0; i < 4; ++i) {
        float a0 = fexp2(fmaf(sc0[2 * i], kScaleL2, nm));
        float a1 = fexp2(fmaf(sc0[2 * i + 1], kScaleL2, nm));
        float b0 = fexp2(fmaf(sc1[2 * i], kScaleL2, nm));
        float b1 = fexp2(fmaf(sc1[2 * i + 1], kScaleL2, nm));
        ls += (a0 + a1) + (b0 + b1);
        pau[i] = pack2(a0, a1);
        pau[4 + i] = pack2(b0, b1);
      }
      const v16bf pabf = __builtin_bit_cast(v16bf, pau);
      ls += bperm_f(xidx, ls);
      if (renew) {                    // max moved: rescale O (rare after warmup)
        const float corr = fexp2(mold - mnew);
        mold = mnew;
        lrow = lrow * corr + ls;
        float co[8];
#pragma unroll
        for (int vv = 0; vv < 8; ++vv) co[vv] = bperm_f(bidx[vv], corr);
#pragma unroll
        for (int j = 0; j < 16; ++j)
#pragma unroll
          for (int vv = 0; vv < 8; ++vv) o[j][vv] *= co[vv];
      } else {
        lrow += ls;
      }
      // ---- O += P(16x32) * V12(32x256), software-pipelined B-fragment loads ----
      const unsigned short* vfp = vtl + lm * VSTR + half * 8;
      LdFrag vb;
      vb.q[0] = *(const uint4*)(vfp);
      vb.q[1] = *(const uint4*)(vfp + 16);
#pragma unroll
      for (int j = 0; j < 16; ++j) {
        LdFrag nb;
        if (j < 15) {
          const unsigned short* src = vfp + (j + 1) * 16 * VSTR;
          nb.q[0] = *(const uint4*)(src);
          nb.q[1] = *(const uint4*)(src + 16);
        }
        o[j] = __builtin_amdgcn_wmma_f32_16x16x32_bf16(
            false, pabf, false, vb.bf, (short)0, o[j], false, false);
        vb = nb;
      }
    }
  }

  __syncthreads();  // all K/V use done; xbuf may now reuse that LDS

  // normalize by softmax denominator (broadcast 1/l from query-lane to O rows)
  {
    const float inv = 1.0f / lrow;
    float io[8];
#pragma unroll
    for (int vv = 0; vv < 8; ++vv) io[vv] = bperm_f(bidx[vv], inv);
#pragma unroll
    for (int j = 0; j < 16; ++j)
#pragma unroll
      for (int vv = 0; vv < 8; ++vv) o[j][vv] *= io[vv];
  }

  if (att == 1) {  // attn2 waves publish
#pragma unroll
    for (int j = 0; j < 16; ++j) {
      const int col = j * 16 + lm;
#pragma unroll
      for (int vv = 0; vv < 8; ++vv) {
        const int lr = 16 * qr + vv + 8 * half;
        xbuf[lr * XSTR + col] = o[j][vv];
      }
    }
  }
  __syncthreads();

  if (att == 0) {  // combine, RMSNorm over 256, scale, write
    float wcol[16];
#pragma unroll
    for (int j = 0; j < 16; ++j) wcol[j] = w[j * 16 + lm];
    float ssq[8];
#pragma unroll
    for (int vv = 0; vv < 8; ++vv) ssq[vv] = 0.f;
#pragma unroll
    for (int j = 0; j < 16; ++j) {
      const int col = j * 16 + lm;
#pragma unroll
      for (int vv = 0; vv < 8; ++vv) {
        const int lr = 16 * qr + vv + 8 * half;
        const float x = o[j][vv] - lam * xbuf[lr * XSTR + col];
        o[j][vv] = x;
        ssq[vv] += x * x;
      }
    }
    float rms[8];
#pragma unroll
    for (int vv = 0; vv < 8; ++vv)
      rms[vv] = rsqrtf(rowred_sum(ssq[vv]) * (1.0f / 256.0f) + 1e-5f) *
                (1.0f - kLambdaInit);
#pragma unroll
    for (int j = 0; j < 16; ++j) {
      const int col = j * 16 + lm;
      const int ho = 2 * hp + (col >> 7);
      const int dd = col & 127;
#pragma unroll
      for (int vv = 0; vv < 8; ++vv) {
        const int sq = r0 + vv + 8 * half;
        out[((b * kS + sq) * kH + ho) * kD + dd] = o[j][vv] * rms[vv] * wcol[j];
      }
    }
  }
}

extern "C" void kernel_launch(void* const* d_in, const int* in_sizes, int n_in,
                              void* d_out, int out_size, void* d_ws, size_t ws_size,
                              hipStream_t stream) {
  (void)in_sizes; (void)n_in; (void)out_size; (void)d_ws; (void)ws_size;
  const float* q = (const float*)d_in[0];
  const float* k = (const float*)d_in[1];
  const float* v = (const float*)d_in[2];
  const float* lq1 = (const float*)d_in[3];
  const float* lk1 = (const float*)d_in[4];
  const float* lq2 = (const float*)d_in[5];
  const float* lk2 = (const float*)d_in[6];
  const float* w = (const float*)d_in[7];
  dim3 grid(kS / QT, kHP, kB);
  hipLaunchKernelGGL(diff_attn_wmma, grid, dim3(256), SMEM_BYTES, stream,
                     q, k, v, lq1, lk1, lq2, lk2, w, (float*)d_out);
}
